// AutopoieticAttention_86053964743133
// MI455X (gfx1250) — matrix operand order repaired and checked by reading, not compiled, not run on hardware
//
#include <hip/hip_runtime.h>
#include <hip/hip_bf16.h>
#include <math.h>

typedef __bf16 bf16_t;
typedef __attribute__((ext_vector_type(8)))  bf16_t v8bf;
typedef __attribute__((ext_vector_type(16))) bf16_t v16bf;
typedef __attribute__((ext_vector_type(8)))  float  v8f;

#define S_LEN 512
#define EMB   512
#define NB    4
#define NH    8
#define HD    64

__device__ __forceinline__ unsigned short f2bf(float f) {
  unsigned int u = __float_as_uint(f);
  u += 0x7FFFu + ((u >> 16) & 1u);
  return (unsigned short)(u >> 16);
}

// ---------------- f32 -> bf16 conversion ----------------
__global__ void k_f32_to_bf16(const float* __restrict__ in,
                              unsigned short* __restrict__ out, int n) {
  int i = blockIdx.x * blockDim.x + threadIdx.x;
  if (i < n) out[i] = f2bf(in[i]);
}

// ---------------- WMMA 32x64 strip helper ----------------
// A: row-major [32, K] (lda elements), B: row-major [K, >=64] (ldb elements).
// 8 accumulators: acc[nt] = rows 0..15, acc[4+nt] = rows 16..31 (cols nt*16..).
// Each B fragment is reused by both A fragments (2x arithmetic intensity).
__device__ __forceinline__ void wmma_strip32x64(const bf16_t* __restrict__ A, int lda,
                                                const bf16_t* __restrict__ B, int ldb,
                                                int K, v8f acc[8]) {
  const int lane = threadIdx.x & 31;
  const int half = lane >> 4;
  const int lr   = lane & 15;
  const bf16_t* ar0 = A + (size_t)lr * lda;
  const bf16_t* ar1 = A + (size_t)(lr + 16) * lda;
  for (int k0 = 0; k0 < K; k0 += 32) {
    // A fragments: lane holds K = k0 + 8*half + [0..7] and k0 + 16 + 8*half + [0..7]
    v8bf a0lo = *(const v8bf*)(ar0 + k0 + half * 8);
    v8bf a0hi = *(const v8bf*)(ar0 + k0 + 16 + half * 8);
    v16bf af0 = __builtin_shufflevector(a0lo, a0hi,
        0,1,2,3,4,5,6,7,8,9,10,11,12,13,14,15);
    v8bf a1lo = *(const v8bf*)(ar1 + k0 + half * 8);
    v8bf a1hi = *(const v8bf*)(ar1 + k0 + 16 + half * 8);
    v16bf af1 = __builtin_shufflevector(a1lo, a1hi,
        0,1,2,3,4,5,6,7,8,9,10,11,12,13,14,15);
    // B fragment: lane holds column nt*16 + lr, K = k0 + 8*half + [0..7] and k0 + 16 + 8*half + [0..7]
    const bf16_t* brow = B + (size_t)k0 * ldb + lr;
#pragma unroll
    for (int nt = 0; nt < 4; ++nt) {
      v16bf bfm;
#pragma unroll
      for (int s = 0; s < 8; ++s) {
        bfm[s]     = brow[(size_t)(half * 8 + s) * ldb + nt * 16];
        bfm[8 + s] = brow[(size_t)(16 + half * 8 + s) * ldb + nt * 16];
      }
      acc[nt]     = __builtin_amdgcn_wmma_f32_16x16x32_bf16(
          false, af0, false, bfm, (short)0, acc[nt],     false, false);
      acc[4 + nt] = __builtin_amdgcn_wmma_f32_16x16x32_bf16(
          false, af1, false, bfm, (short)0, acc[4 + nt], false, false);
    }
  }
}

// ---------------- QKV projection GEMM: [2048,512] x [512,512] + bias ----------------
// ktmode==0: store bf16 row-major [M, 512]
// ktmode==1: scatter store K transposed as [B, H, hd, S] bf16
__global__ void k_gemm_proj(const unsigned short* __restrict__ Aus,
                            const unsigned short* __restrict__ Bus,
                            const float* __restrict__ bias,
                            unsigned short* __restrict__ out, int ktmode) {
  const int wave = threadIdx.x >> 5;
  const int lane = threadIdx.x & 31;
  const int half = lane >> 4, lr = lane & 15;
  const int m0 = (blockIdx.y * 4 + wave) * 32;
  const int n0 = blockIdx.x * 64;
  v8f acc[8] = {};
  wmma_strip32x64((const bf16_t*)Aus + (size_t)m0 * EMB, EMB,
                  (const bf16_t*)Bus + n0, EMB, EMB, acc);
#pragma unroll
  for (int mt = 0; mt < 2; ++mt) {
#pragma unroll
    for (int nt = 0; nt < 4; ++nt) {
      const int n = n0 + nt * 16 + lr;
      const float bv = bias[n];
#pragma unroll
      for (int i = 0; i < 8; ++i) {
        const int m = m0 + mt * 16 + half * 8 + i;
        const float val = acc[mt * 4 + nt][i] + bv;
        if (ktmode) {
          const int b = m >> 9, srow = m & (S_LEN - 1);
          const int h = n >> 6, d = n & (HD - 1);
          out[((size_t)((b * NH + h) * HD + d)) * S_LEN + srow] = f2bf(val);
        } else {
          out[(size_t)m * EMB + n] = f2bf(val);
        }
      }
    }
  }
}

// ---------------- scores: s[b,h] = clip(nan_to_num(q @ k^T * scale)) ----------------
__global__ void k_gemm_scores(const unsigned short* __restrict__ qus,
                              const unsigned short* __restrict__ ktus,
                              float* __restrict__ sout) {
  const int wave = threadIdx.x >> 5;
  const int lane = threadIdx.x & 31;
  const int half = lane >> 4, lr = lane & 15;
  const int z = blockIdx.z;
  const int b = z >> 3, h = z & 7;
  const int m0 = (blockIdx.y * 4 + wave) * 32;
  const int n0 = blockIdx.x * 64;
  v8f acc[8] = {};
  const bf16_t* A = (const bf16_t*)qus + ((size_t)(b * S_LEN + m0)) * EMB + h * HD;
  const bf16_t* B = (const bf16_t*)ktus + ((size_t)(b * NH + h) * HD) * S_LEN + n0;
  wmma_strip32x64(A, EMB, B, S_LEN, HD, acc);
  const float scale = 0.125f;  // 1/sqrt(64)
  float* sz = sout + (size_t)z * S_LEN * S_LEN;
#pragma unroll
  for (int mt = 0; mt < 2; ++mt) {
#pragma unroll
    for (int nt = 0; nt < 4; ++nt) {
#pragma unroll
      for (int i = 0; i < 8; ++i) {
        float v = acc[mt * 4 + nt][i] * scale;
        if (__builtin_isnan(v)) v = 0.f;
        else if (__builtin_isinf(v)) v = copysignf(10.f, v);
        v = fminf(fmaxf(v, -15.f), 15.f);
        sz[(size_t)(m0 + mt * 16 + half * 8 + i) * S_LEN + n0 + nt * 16 + lr] = v;
      }
    }
  }
}

// ---------------- per-row autopoietic transform ----------------
// One block (256 threads) per (b, row i): computes ma row, p-softmax, entropy,
// Fm-softmax, 128-channel scalar MLP, tprime row, and 6 per-row partials.
__global__ void k_transform(const float* __restrict__ sbuf,
                            const float* __restrict__ w1, const float* __restrict__ b1,
                            const float* __restrict__ w2, const float* __restrict__ b2,
                            float* __restrict__ mabuf, float* __restrict__ tpbuf,
                            float* __restrict__ rowpart) {
  __shared__ float w1s[128], b1s[128], w2s[128];
  __shared__ float red[256];
  const int t = threadIdx.x;
  const int row = blockIdx.x;          // 0..B*S-1
  const int b = row >> 9, i = row & (S_LEN - 1);
  if (t < 128) { w1s[t] = w1[t]; b1s[t] = b1[t]; w2s[t] = w2[t]; }

  float ml[2];
#pragma unroll
  for (int jj = 0; jj < 2; ++jj) {
    const int j = t + jj * 256;
    float a = 0.f;
#pragma unroll
    for (int h = 0; h < NH; ++h)
      a += sbuf[(((size_t)(b * NH + h)) * S_LEN + i) * S_LEN + j];
    const float ma = a * 0.125f;
    ml[jj] = ma;
    mabuf[(size_t)row * S_LEN + j] = ma;
  }
  __syncthreads();

  // p = softmax(clip(ma,-10,10)); Hent = -p*log(p+1e-6)
  const float c0 = fminf(fmaxf(ml[0], -10.f), 10.f);
  const float c1 = fminf(fmaxf(ml[1], -10.f), 10.f);
  red[t] = fmaxf(c0, c1); __syncthreads();
  for (int st = 128; st > 0; st >>= 1) { if (t < st) red[t] = fmaxf(red[t], red[t + st]); __syncthreads(); }
  const float mx = red[0]; __syncthreads();
  float e0 = expf(c0 - mx), e1 = expf(c1 - mx);
  red[t] = e0 + e1; __syncthreads();
  for (int st = 128; st > 0; st >>= 1) { if (t < st) red[t] += red[t + st]; __syncthreads(); }
  const float Z = red[0]; __syncthreads();
  const float p0 = e0 / Z, p1 = e1 / Z;
  const float he0 = -p0 * logf(p0 + 1e-6f);
  const float he1 = -p1 * logf(p1 + 1e-6f);

  // Fm = softmax(3*Hent)
  red[t] = fmaxf(3.f * he0, 3.f * he1); __syncthreads();
  for (int st = 128; st > 0; st >>= 1) { if (t < st) red[t] = fmaxf(red[t], red[t + st]); __syncthreads(); }
  const float mx2 = red[0]; __syncthreads();
  float f0 = expf(3.f * he0 - mx2), f1 = expf(3.f * he1 - mx2);
  red[t] = f0 + f1; __syncthreads();
  for (int st = 128; st > 0; st >>= 1) { if (t < st) red[t] += red[t + st]; __syncthreads(); }
  const float Z2 = red[0]; __syncthreads();
  f0 /= Z2; f1 /= Z2;

  // 128-channel scalar MLP -> sigmoid gate; tprime = sig * Fm
  const float b2v = b2[0];
  float tp[2];
  const float fm[2] = {f0, f1};
#pragma unroll
  for (int jj = 0; jj < 2; ++jj) {
    const int j = t + jj * 256;
    const float sa = fminf(fmaxf(ml[jj], -8.f), 8.f) * 0.05f;
    float apacc = 0.f;
    for (int c = 0; c < 128; ++c) {
      float u = fminf(fmaxf(sa * w1s[c] + b1s[c], -5.f), 5.f);
      u = fmaxf(u, 0.f);
      apacc += u * w2s[c];
    }
    const float ap = fminf(fmaxf(apacc + b2v, -5.f), 5.f);
    const float ty = fminf(fmaxf(1.f + 2.5f * ap, 0.5f), 1.5f);
    const float sg = 1.f / (1.f + expf(-ty));
    tp[jj] = sg * fm[jj];
    tpbuf[(size_t)row * S_LEN + j] = tp[jj];
  }

  // per-row partials: {sum ma, sum ma^2, max|ma|, sum tp, sum tp^2, sum Hent}
  float part[6];
  part[0] = ml[0] + ml[1];
  part[1] = ml[0] * ml[0] + ml[1] * ml[1];
  part[2] = fmaxf(fabsf(ml[0]), fabsf(ml[1]));
  part[3] = tp[0] + tp[1];
  part[4] = tp[0] * tp[0] + tp[1] * tp[1];
  part[5] = he0 + he1;
#pragma unroll
  for (int q = 0; q < 6; ++q) {
    red[t] = part[q]; __syncthreads();
    if (q == 2) {
      for (int st = 128; st > 0; st >>= 1) { if (t < st) red[t] = fmaxf(red[t], red[t + st]); __syncthreads(); }
    } else {
      for (int st = 128; st > 0; st >>= 1) { if (t < st) red[t] += red[t + st]; __syncthreads(); }
    }
    if (t == 0) rowpart[(size_t)row * 8 + q] = red[0];
    __syncthreads();
  }
}

// ---------------- per-batch reduction -> blend coefficients {alpha, rr, gamma} ----------------
__global__ void k_batch_reduce(const float* __restrict__ rowpart,
                               float* __restrict__ bscal) {
  __shared__ float red[256];
  const int t = threadIdx.x, b = blockIdx.x;
  const float* rp = rowpart + (size_t)b * S_LEN * 8;
  float loc[6];
#pragma unroll
  for (int q = 0; q < 6; ++q) {
    const float a = rp[(size_t)t * 8 + q];
    const float c = rp[(size_t)(t + 256) * 8 + q];
    loc[q] = (q == 2) ? fmaxf(a, c) : (a + c);
  }
  float tot[6];
#pragma unroll
  for (int q = 0; q < 6; ++q) {
    red[t] = loc[q]; __syncthreads();
    if (q == 2) {
      for (int st = 128; st > 0; st >>= 1) { if (t < st) red[t] = fmaxf(red[t], red[t + st]); __syncthreads(); }
    } else {
      for (int st = 128; st > 0; st >>= 1) { if (t < st) red[t] += red[t + st]; __syncthreads(); }
    }
    tot[q] = red[0]; __syncthreads();
  }
  if (t == 0) {
    const float Nf = (float)S_LEN * (float)S_LEN;
    const float eo = sqrtf(tot[1]) + 1e-4f;
    const float et = sqrtf(tot[4]) + 1e-4f;
    const float c  = fminf(fmaxf(eo / et, 0.8f), 1.2f);
    const float tpm = tot[3] / Nf;              // mean tprime
    const float tm  = c * tpm;                  // mean t
    const float omn = tot[0] / Nf;
    const float tvar = fmaxf(c * c * fmaxf(tot[4] / Nf - tpm * tpm, 0.f), 0.01f);
    const float ovar = fmaxf(tot[1] / Nf - omn * omn, 0.01f);
    const float gd = fminf(fmaxf(sqrtf(ovar) / sqrtf(tvar), 0.8f), 1.2f);
    const float ar = fminf(fmaxf(tot[2], 1.f), 10.f);
    const float sm = fminf(fmaxf(0.3f / log1pf(ar), 0.1f), 0.5f);
    const float ent = tot[5] / Nf;
    const float ne = ent / logf((float)S_LEN);
    const float rr = 0.4f * (1.f - fminf(fmaxf(ne, 0.f), 0.4f));
    const float smgd = sm * gd;
    bscal[b * 4 + 0] = rr * smgd * c;           // alpha (coef for tprime)
    bscal[b * 4 + 1] = rr;                      // coef for ma (negated)
    bscal[b * 4 + 2] = rr * tm * (1.f - smgd);  // gamma
    bscal[b * 4 + 3] = 0.f;
  }
}

// ---------------- adjusted softmax -> bf16 attn ----------------
__global__ void k_softmax_attn(const float* __restrict__ sbuf,
                               const float* __restrict__ mabuf,
                               const float* __restrict__ tpbuf,
                               const float* __restrict__ bscal,
                               const float* __restrict__ tau,
                               unsigned short* __restrict__ attn) {
  __shared__ float red[256];
  const int t = threadIdx.x;
  const int ridx = blockIdx.x;       // 0..B*H*S-1
  const int i = ridx & (S_LEN - 1);
  const int bh = ridx >> 9;
  const int b = bh >> 3;
  const float alpha = bscal[b * 4 + 0];
  const float rr    = bscal[b * 4 + 1];
  const float gam   = bscal[b * 4 + 2];
  const float invtau = 1.f / tau[0];
  const float* sr = sbuf + (size_t)ridx * S_LEN;
  const float* mr = mabuf + ((size_t)(b * S_LEN + i)) * S_LEN;
  const float* tr = tpbuf + ((size_t)(b * S_LEN + i)) * S_LEN;
  const float v0 = (sr[t]       + alpha * tr[t]       - rr * mr[t]       + gam) * invtau;
  const float v1 = (sr[t + 256] + alpha * tr[t + 256] - rr * mr[t + 256] + gam) * invtau;
  red[t] = fmaxf(v0, v1); __syncthreads();
  for (int st = 128; st > 0; st >>= 1) { if (t < st) red[t] = fmaxf(red[t], red[t + st]); __syncthreads(); }
  const float mx = red[0]; __syncthreads();
  const float e0 = expf(v0 - mx), e1 = expf(v1 - mx);
  red[t] = e0 + e1; __syncthreads();
  for (int st = 128; st > 0; st >>= 1) { if (t < st) red[t] += red[t + st]; __syncthreads(); }
  const float invZ = 1.f / red[0];
  attn[(size_t)ridx * S_LEN + t]       = f2bf(e0 * invZ);
  attn[(size_t)ridx * S_LEN + t + 256] = f2bf(e1 * invZ);
}

// ---------------- attn @ v -> merged heads O (bf16 [B,S,E]) ----------------
__global__ void k_gemm_av(const unsigned short* __restrict__ attn,
                          const unsigned short* __restrict__ vus,
                          unsigned short* __restrict__ obuf) {
  const int wave = threadIdx.x >> 5;
  const int lane = threadIdx.x & 31;
  const int half = lane >> 4, lr = lane & 15;
  const int z = blockIdx.z;
  const int b = z >> 3, h = z & 7;
  const int m0 = (blockIdx.y * 4 + wave) * 32;
  v8f acc[8] = {};
  const bf16_t* A = (const bf16_t*)attn + ((size_t)z * S_LEN + m0) * S_LEN;
  const bf16_t* B = (const bf16_t*)vus + (size_t)b * S_LEN * EMB + h * HD;
  wmma_strip32x64(A, S_LEN, B, EMB, S_LEN, acc);
#pragma unroll
  for (int mt = 0; mt < 2; ++mt) {
#pragma unroll
    for (int nt = 0; nt < 4; ++nt) {
#pragma unroll
      for (int i = 0; i < 8; ++i) {
        obuf[((size_t)(b * S_LEN + m0 + mt * 16 + half * 8 + i)) * EMB +
             h * HD + nt * 16 + lr] = f2bf(acc[mt * 4 + nt][i]);
      }
    }
  }
}

// ---------------- final: O @ Wo + bo -> f32 out ----------------
__global__ void k_gemm_final(const unsigned short* __restrict__ ous,
                             const unsigned short* __restrict__ wous,
                             const float* __restrict__ bo,
                             float* __restrict__ out) {
  const int wave = threadIdx.x >> 5;
  const int lane = threadIdx.x & 31;
  const int half = lane >> 4, lr = lane & 15;
  const int m0 = (blockIdx.y * 4 + wave) * 32;
  const int n0 = blockIdx.x * 64;
  v8f acc[8] = {};
  wmma_strip32x64((const bf16_t*)ous + (size_t)m0 * EMB, EMB,
                  (const bf16_t*)wous + n0, EMB, EMB, acc);
#pragma unroll
  for (int mt = 0; mt < 2; ++mt) {
#pragma unroll
    for (int nt = 0; nt < 4; ++nt) {
      const int n = n0 + nt * 16 + lr;
      const float bv = bo[n];
#pragma unroll
      for (int i = 0; i < 8; ++i)
        out[(size_t)(m0 + mt * 16 + half * 8 + i) * EMB + n] = acc[mt * 4 + nt][i] + bv;
    }
  }
}

extern "C" void kernel_launch(void* const* d_in, const int* in_sizes, int n_in,
                              void* d_out, int out_size, void* d_ws, size_t ws_size,
                              hipStream_t stream) {
  const float* x   = (const float*)d_in[0];
  const float* Wq  = (const float*)d_in[1];
  const float* bq  = (const float*)d_in[2];
  const float* Wk  = (const float*)d_in[3];
  const float* bk  = (const float*)d_in[4];
  const float* Wv  = (const float*)d_in[5];
  const float* bv  = (const float*)d_in[6];
  const float* Wo  = (const float*)d_in[7];
  const float* bo  = (const float*)d_in[8];
  const float* w1  = (const float*)d_in[9];
  const float* b1  = (const float*)d_in[10];
  const float* w2  = (const float*)d_in[11];
  const float* b2  = (const float*)d_in[12];
  const float* tau = (const float*)d_in[13];
  float* out = (float*)d_out;

  const size_t M = (size_t)NB * S_LEN;                 // 2048
  char* p = (char*)d_ws;
  auto carve = [&](size_t bytes) -> void* {
    void* r = (void*)p; p += (bytes + 255) & ~(size_t)255; return r;
  };
  unsigned short* xb   = (unsigned short*)carve(M * EMB * 2);
  unsigned short* wqb  = (unsigned short*)carve((size_t)EMB * EMB * 2);
  unsigned short* wkb  = (unsigned short*)carve((size_t)EMB * EMB * 2);
  unsigned short* wvb  = (unsigned short*)carve((size_t)EMB * EMB * 2);
  unsigned short* wob  = (unsigned short*)carve((size_t)EMB * EMB * 2);
  unsigned short* qb   = (unsigned short*)carve(M * EMB * 2);
  unsigned short* ktb  = (unsigned short*)carve((size_t)NB * NH * HD * S_LEN * 2);
  unsigned short* vbuf = (unsigned short*)carve(M * EMB * 2);
  float* sbuf          = (float*)carve((size_t)NB * NH * S_LEN * S_LEN * 4);
  float* mabuf         = (float*)carve((size_t)NB * S_LEN * S_LEN * 4);
  float* tpbuf         = (float*)carve((size_t)NB * S_LEN * S_LEN * 4);
  float* rowpart       = (float*)carve((size_t)NB * S_LEN * 8 * 4);
  float* bscal         = (float*)carve((size_t)NB * 4 * 4);
  unsigned short* attn = (unsigned short*)carve((size_t)NB * NH * S_LEN * S_LEN * 2);
  unsigned short* obuf = (unsigned short*)carve(M * EMB * 2);

  // 1) convert activations + weights to bf16
  const int nx = (int)(M * EMB), nw = EMB * EMB;
  k_f32_to_bf16<<<(nx + 255) / 256, 256, 0, stream>>>(x,  xb,  nx);
  k_f32_to_bf16<<<(nw + 255) / 256, 256, 0, stream>>>(Wq, wqb, nw);
  k_f32_to_bf16<<<(nw + 255) / 256, 256, 0, stream>>>(Wk, wkb, nw);
  k_f32_to_bf16<<<(nw + 255) / 256, 256, 0, stream>>>(Wv, wvb, nw);
  k_f32_to_bf16<<<(nw + 255) / 256, 256, 0, stream>>>(Wo, wob, nw);

  // 2) Q/K/V projections (WMMA, 32x64 wave tiles). K stored transposed [B,H,hd,S].
  dim3 gProj(EMB / 64, (unsigned)(M / 128));
  k_gemm_proj<<<gProj, 128, 0, stream>>>(xb, wqb, bq, qb,   0);
  k_gemm_proj<<<gProj, 128, 0, stream>>>(xb, wkb, bk, ktb,  1);
  k_gemm_proj<<<gProj, 128, 0, stream>>>(xb, wvb, bv, vbuf, 0);

  // 3) scores s = clip(nan_to_num(q k^T / sqrt(hd)))  (WMMA)
  dim3 gScores(S_LEN / 64, S_LEN / 128, NB * NH);
  k_gemm_scores<<<gScores, 128, 0, stream>>>(qb, ktb, sbuf);

  // 4) per-row autopoietic transform + per-row partials
  k_transform<<<NB * S_LEN, 256, 0, stream>>>(sbuf, w1, b1, w2, b2,
                                              mabuf, tpbuf, rowpart);

  // 5) per-batch reduction -> blend coefficients
  k_batch_reduce<<<NB, 256, 0, stream>>>(rowpart, bscal);

  // 6) fused blend + softmax -> bf16 attention
  k_softmax_attn<<<NB * NH * S_LEN, 256, 0, stream>>>(sbuf, mabuf, tpbuf,
                                                      bscal, tau, attn);

  // 7) attn @ v -> merged-head O (WMMA)
  dim3 gAV(1, S_LEN / 128, NB * NH);
  k_gemm_av<<<gAV, 128, 0, stream>>>(attn, vbuf, obuf);

  // 8) O @ Wo + bo -> f32 output (WMMA)
  k_gemm_final<<<gProj, 128, 0, stream>>>(obuf, wob, bo, out);

  (void)in_sizes; (void)n_in; (void)out_size; (void)ws_size;
}
